// BidirectionalCrossModalScan_70738111365361
// MI455X (gfx1250) — compile-verified, matrix operand order checked by reading, not run on hardware
//
#include <hip/hip_runtime.h>
#include <hip/hip_bf16.h>
#include <math.h>

// ---------------- problem constants (from reference) ----------------
#define DIM      192
#define NLAYERS  4
#define STATE    16
#define HEADDIM  64
#define DIN      384            // EXPAND*DIM
#define NHEADS   6
#define CONVK    4
#define CONVCH   416            // DIN + 2*STATE
#define PROJ     806            // 2*DIN + 2*STATE + NHEADS
#define PROJP    832            // PROJ padded to multiple of 32 (for 2-wide N tiling)
#define BATCH    8
#define HALFSEQ  4096
#define SEQ      8192           // interleaved length 2*N
#define MROWS    (BATCH * SEQ)  // 65536 token rows

// register-tile shape per wave: 4 M-tiles x 2 N-tiles of 16x16
#define MT 4
#define NT 2

typedef __bf16 bf16_t;
typedef __attribute__((ext_vector_type(16))) __bf16 v16bf;
typedef __attribute__((ext_vector_type(8)))  __bf16 v8bf;
typedef __attribute__((ext_vector_type(8)))  float  v8f;

__device__ __forceinline__ bf16_t to_bf16(float f) {
  union { float f; unsigned u; } v; v.f = f;
  unsigned r = v.u + 0x7FFFu + ((v.u >> 16) & 1u);   // round-to-nearest-even
  unsigned short h = (unsigned short)(r >> 16);
  bf16_t out;
  __builtin_memcpy(&out, &h, 2);
  return out;
}

// --------------- WMMA fragment loader (bf16 16x32, A layout) ---------------
// lanes 0-15: M=lane, K runs {k0..k0+7, k0+16..k0+23}
// lanes 16-31: M=lane-16, K runs {k0+8..k0+15, k0+24..k0+31}
// Two contiguous 16-byte loads per lane. B fragment = same loader on W^T.
__device__ __forceinline__ v16bf load_frag(const bf16_t* __restrict__ X, int ld,
                                           int row0, int k0, int lane) {
  const int r  = lane & 15;
  const int hi = lane >> 4;
  const bf16_t* p = X + (size_t)(row0 + r) * ld + (k0 + hi * 8);
  v8bf lo  = *(const v8bf*)p;
  v8bf hi8 = *(const v8bf*)(p + 16);
  v16bf f;
#pragma unroll
  for (int i = 0; i < 8; ++i) { f[i] = lo[i]; f[i + 8] = hi8[i]; }
  return f;
}

// 4x2 register-tiled mainloop: 6 fragment loads feed 8 WMMAs per K-step.
__device__ __forceinline__ void wmma_tile_4x2(const bf16_t* __restrict__ X, int ldx,
                                              const bf16_t* __restrict__ Wt, int ldw,
                                              int m0, int n0, int K, int lane,
                                              v8f acc[MT][NT]) {
  for (int k0 = 0; k0 < K; k0 += 32) {
    v16bf a[MT], b[NT];
#pragma unroll
    for (int i = 0; i < MT; ++i) a[i] = load_frag(X, ldx, m0 + 16 * i, k0, lane);
#pragma unroll
    for (int j = 0; j < NT; ++j) b[j] = load_frag(Wt, ldw, n0 + 16 * j, k0, lane);
#pragma unroll
    for (int i = 0; i < MT; ++i)
#pragma unroll
      for (int j = 0; j < NT; ++j)
        acc[i][j] = __builtin_amdgcn_wmma_f32_16x16x32_bf16(
            false, a[i], false, b[j], (short)0, acc[i][j], false, false);
  }
}

// ---------------- prep kernels ----------------
// interleave [s,f,s,f,...] into Hf; reversed copy into Hb
__global__ void k_interleave(const float* __restrict__ sem, const float* __restrict__ feat,
                             float* __restrict__ Hf, float* __restrict__ Hb) {
  int idx = blockIdx.x * blockDim.x + threadIdx.x;   // MROWS*DIM
  int m = idx / DIM, c = idx % DIM;
  int b = m >> 13, l = m & (SEQ - 1), n = l >> 1;
  const float* src = (l & 1) ? feat : sem;
  float val = src[((size_t)b * HALFSEQ + n) * DIM + c];
  Hf[idx] = val;
  Hb[(((size_t)b * SEQ) + (SEQ - 1 - l)) * DIM + c] = val;
}

// in_w (192 x 806) -> bf16 W^T padded (832 x 192)
__global__ void k_prep_inw(const float* __restrict__ W, bf16_t* __restrict__ Wt) {
  int idx = blockIdx.x * blockDim.x + threadIdx.x;   // PROJP*DIM
  int n = idx / DIM, k = idx % DIM;
  float v = (n < PROJ) ? W[(size_t)k * PROJ + n] : 0.f;
  Wt[idx] = to_bf16(v);
}

// W (384 x N) -> bf16 W^T (N x 384). Used for out_w and gate_w (N = 192).
__global__ void k_prep_w384(const float* __restrict__ W, int N, bf16_t* __restrict__ Wt) {
  int idx = blockIdx.x * blockDim.x + threadIdx.x;   // N*DIN
  int n = idx / DIN, k = idx % DIN;
  Wt[idx] = to_bf16(W[(size_t)k * N + n]);
}

// rmsnorm: one wave per row of 192, writes bf16 normalized activations
__global__ void k_rmsnorm(const float* __restrict__ H, const float* __restrict__ w,
                          bf16_t* __restrict__ Xn) {
  const int lane = threadIdx.x & 31;
  const int row  = (blockIdx.x * blockDim.x + threadIdx.x) >> 5;
  const float* x = H + (size_t)row * DIM;
  float v[6]; float ss = 0.f;
#pragma unroll
  for (int i = 0; i < 6; ++i) { v[i] = x[lane + 32 * i]; ss += v[i] * v[i]; }
#pragma unroll
  for (int m = 16; m; m >>= 1) ss += __shfl_xor(ss, m, 32);
  const float scale = rsqrtf(ss * (1.f / DIM) + 1e-5f);
#pragma unroll
  for (int i = 0; i < 6; ++i) {
    int c = lane + 32 * i;
    Xn[(size_t)row * DIM + c] = to_bf16(v[i] * scale * w[c]);
  }
}

// ---------------- WMMA GEMMs (4x2 register tiles per wave) ----------------
// zxbcdt = Xn(65536x192) @ in_w + in_b  -> f32 (65536 x 832 padded)
__global__ void k_gemm_inproj(const bf16_t* __restrict__ Xn, const bf16_t* __restrict__ Wt,
                              const float* __restrict__ bias, float* __restrict__ Out) {
  const int lane = threadIdx.x & 31;
  const int wid  = (blockIdx.x * blockDim.x + threadIdx.x) >> 5;
  const int nblk = PROJP / (16 * NT);            // 26
  const int m0 = (wid / nblk) * (16 * MT);
  const int n0 = (wid % nblk) * (16 * NT);
  v8f acc[MT][NT] = {};
  wmma_tile_4x2(Xn, DIM, Wt, DIM, m0, n0, DIM, lane, acc);
  const int hi8 = (lane >> 4) << 3;
#pragma unroll
  for (int j = 0; j < NT; ++j) {
    const int col = n0 + 16 * j + (lane & 15);
    const float b = (col < PROJ) ? bias[col] : 0.f;
#pragma unroll
    for (int i = 0; i < MT; ++i) {
      const int rbase = m0 + 16 * i + hi8;
#pragma unroll
      for (int r = 0; r < 8; ++r)
        Out[(size_t)(rbase + r) * PROJP + col] = acc[i][j][r] + b;
    }
  }
}

// H += Yb(65536x384) @ out_w + out_b (residual in-place)
__global__ void k_gemm_outproj(const bf16_t* __restrict__ Yb, const bf16_t* __restrict__ Wt,
                               const float* __restrict__ bias, float* __restrict__ H) {
  const int lane = threadIdx.x & 31;
  const int wid  = (blockIdx.x * blockDim.x + threadIdx.x) >> 5;
  const int nblk = DIM / (16 * NT);              // 6
  const int m0 = (wid / nblk) * (16 * MT);
  const int n0 = (wid % nblk) * (16 * NT);
  v8f acc[MT][NT] = {};
  wmma_tile_4x2(Yb, DIN, Wt, DIN, m0, n0, DIN, lane, acc);
  const int hi8 = (lane >> 4) << 3;
#pragma unroll
  for (int j = 0; j < NT; ++j) {
    const int col = n0 + 16 * j + (lane & 15);
    const float b = bias[col];
#pragma unroll
    for (int i = 0; i < MT; ++i) {
      const int rbase = m0 + 16 * i + hi8;
#pragma unroll
      for (int r = 0; r < 8; ++r) {
        size_t o = (size_t)(rbase + r) * DIM + col;
        H[o] += acc[i][j][r] + b;
      }
    }
  }
}

// gate = sigmoid([fwd,bwd] @ gate_w + gate_b); merged = g*fwd + (1-g)*bwd; deinterleave
__global__ void k_gemm_gate(const bf16_t* __restrict__ Xg, const bf16_t* __restrict__ Wt,
                            const float* __restrict__ gate_b,
                            const float* __restrict__ Hf, const float* __restrict__ Hb,
                            float* __restrict__ Out) {
  const int lane = threadIdx.x & 31;
  const int wid  = (blockIdx.x * blockDim.x + threadIdx.x) >> 5;
  const int nblk = DIM / (16 * NT);              // 6
  const int m0 = (wid / nblk) * (16 * MT);
  const int n0 = (wid % nblk) * (16 * NT);
  v8f acc[MT][NT] = {};
  wmma_tile_4x2(Xg, 2 * DIM, Wt, 2 * DIM, m0, n0, 2 * DIM, lane, acc);
  const int hi8 = (lane >> 4) << 3;
#pragma unroll
  for (int j = 0; j < NT; ++j) {
    const int col = n0 + 16 * j + (lane & 15);
    const float gb = gate_b[col];
#pragma unroll
    for (int i = 0; i < MT; ++i) {
      const int rbase = m0 + 16 * i + hi8;
#pragma unroll
      for (int r = 0; r < 8; ++r) {
        const int row = rbase + r;
        const int b = row >> 13, l = row & (SEQ - 1);
        const float g = 1.f / (1.f + expf(-(acc[i][j][r] + gb)));
        const float fv = Hf[(size_t)row * DIM + col];
        const float wv = Hb[(((size_t)b * SEQ) + (SEQ - 1 - l)) * DIM + col];
        const float mv = g * fv + (1.f - g) * wv;
        float* dst = Out + ((l & 1) ? (size_t)BATCH * HALFSEQ * DIM : 0);
        dst[((size_t)b * HALFSEQ + (l >> 1)) * DIM + col] = mv;
      }
    }
  }
}

// ---------------- elementwise mid-layer kernels ----------------
// depthwise causal conv K=4 over xbc channels + SiLU
__global__ void k_conv(const float* __restrict__ zx, const float* __restrict__ cw,
                       const float* __restrict__ cb, float* __restrict__ out) {
  int idx = blockIdx.x * blockDim.x + threadIdx.x;   // MROWS*CONVCH
  int m = idx / CONVCH, c = idx % CONVCH;
  int l = m & (SEQ - 1);
  float acc = cb[c];
#pragma unroll
  for (int k = 0; k < CONVK; ++k) {
    int ll = l - (CONVK - 1) + k;
    if (ll >= 0)
      acc += cw[k * CONVCH + c] * zx[(size_t)(m - (CONVK - 1) + k) * PROJP + DIN + c];
  }
  out[idx] = acc / (1.f + expf(-acc));
}

// dt = softplus(dt_raw + bias); dA = exp(dt * -exp(A_log))
__global__ void k_dt(const float* __restrict__ zx, const float* __restrict__ dt_bias,
                     const float* __restrict__ A_log,
                     float* __restrict__ dt, float* __restrict__ dA) {
  int idx = blockIdx.x * blockDim.x + threadIdx.x;   // MROWS*NHEADS
  int m = idx / NHEADS, hh = idx % NHEADS;
  float raw = zx[(size_t)m * PROJP + (DIN + CONVCH) + hh] + dt_bias[hh];
  float d = (raw > 20.f) ? raw : log1pf(expf(raw));
  float A = -expf(A_log[hh]);
  dt[idx] = d;
  dA[idx] = expf(d * A);
}

// sequential SSD scan: one 512-thread block per (batch, head)
// thread t -> p = t/8, two state columns n = 2*(t%8)+{0,1}
__global__ void k_scan(const float* __restrict__ xbc, const float* __restrict__ dt,
                       const float* __restrict__ dA, const float* __restrict__ Dp,
                       float* __restrict__ yraw) {
  const int t  = threadIdx.x;
  const int p  = t >> 3;
  const int nb = (t & 7) << 1;
  const int b  = blockIdx.x / NHEADS;
  const int hh = blockIdx.x % NHEADS;
  const float Dv = Dp[hh];
  float s0 = 0.f, s1 = 0.f;
  size_t m = (size_t)b * SEQ;
  for (int l = 0; l < SEQ; ++l, ++m) {
    const float* row = xbc + m * CONVCH;
    const float x   = row[hh * HEADDIM + p];
    const float B0  = row[DIN + nb];
    const float B1  = row[DIN + nb + 1];
    const float C0  = row[DIN + STATE + nb];
    const float C1  = row[DIN + STATE + nb + 1];
    const float da  = dA[m * NHEADS + hh];
    const float dtv = dt[m * NHEADS + hh];
    const float u = dtv * x;
    s0 = s0 * da + u * B0;
    s1 = s1 * da + u * B1;
    float y = s0 * C0 + s1 * C1;
    y += __shfl_xor(y, 1, 32);
    y += __shfl_xor(y, 2, 32);
    y += __shfl_xor(y, 4, 32);
    if ((t & 7) == 0) yraw[m * DIN + hh * HEADDIM + p] = y + Dv * x;
    if (l + 1 < SEQ) __builtin_prefetch(row + CONVCH, 0, 0);  // global_prefetch_b8
  }
}

// yb = bf16( y * silu(z) )
__global__ void k_ygate(const float* __restrict__ y, const float* __restrict__ zx,
                        bf16_t* __restrict__ yb) {
  int idx = blockIdx.x * blockDim.x + threadIdx.x;   // MROWS*DIN
  int m = idx / DIN, c = idx % DIN;
  float z = zx[(size_t)m * PROJP + c];
  yb[idx] = to_bf16(y[idx] * (z / (1.f + expf(-z))));
}

// build gate input [fwd | unflipped bwd] in bf16
__global__ void k_gateprep(const float* __restrict__ Hf, const float* __restrict__ Hb,
                           bf16_t* __restrict__ Xg) {
  int idx = blockIdx.x * blockDim.x + threadIdx.x;   // MROWS*2*DIM
  int m = idx / (2 * DIM), c = idx % (2 * DIM);
  float v;
  if (c < DIM) {
    v = Hf[(size_t)m * DIM + c];
  } else {
    int b = m >> 13, l = m & (SEQ - 1);
    v = Hb[(((size_t)b * SEQ) + (SEQ - 1 - l)) * DIM + (c - DIM)];
  }
  Xg[idx] = to_bf16(v);
}

// ---------------- host-side orchestration ----------------
extern "C" void kernel_launch(void* const* d_in, const int* /*in_sizes*/, int /*n_in*/,
                              void* d_out, int /*out_size*/, void* d_ws, size_t /*ws_size*/,
                              hipStream_t stream) {
  const float* sem  = (const float*)d_in[0];
  const float* feat = (const float*)d_in[1];
  // params flattened in dict insertion order:
  // per stack: norm_w, in_w, in_b, conv_w, conv_b, dt_bias, A_log, D, out_w, out_b
  struct Stack {
    const float *norm_w, *in_w, *in_b, *conv_w, *conv_b, *dt_bias, *A_log, *D, *out_w, *out_b;
  };
  Stack st[2];
  for (int s = 0; s < 2; ++s) {
    int base = 2 + s * 10;
    st[s].norm_w  = (const float*)d_in[base + 0];
    st[s].in_w    = (const float*)d_in[base + 1];
    st[s].in_b    = (const float*)d_in[base + 2];
    st[s].conv_w  = (const float*)d_in[base + 3];
    st[s].conv_b  = (const float*)d_in[base + 4];
    st[s].dt_bias = (const float*)d_in[base + 5];
    st[s].A_log   = (const float*)d_in[base + 6];
    st[s].D       = (const float*)d_in[base + 7];
    st[s].out_w   = (const float*)d_in[base + 8];
    st[s].out_b   = (const float*)d_in[base + 9];
  }
  const float* gate_w = (const float*)d_in[22];
  const float* gate_b = (const float*)d_in[23];

  // ---- carve scratch ----
  char* ws = (char*)d_ws;
  auto alloc = [&](size_t bytes) -> char* {
    char* p = ws;
    ws += (bytes + 255) & ~(size_t)255;
    return p;
  };
  float*  Hf  = (float*)alloc((size_t)MROWS * DIM * 4);      // fwd activations
  float*  Hb  = (float*)alloc((size_t)MROWS * DIM * 4);      // bwd activations (flipped)
  bf16_t* Xn  = (bf16_t*)alloc((size_t)MROWS * DIM * 2);     // normalized bf16
  float*  ZX  = (float*)alloc((size_t)MROWS * PROJP * 4);    // zxbcdt (padded)
  float*  XC  = (float*)alloc((size_t)MROWS * CONVCH * 4);   // conv+silu output
  float*  DT  = (float*)alloc((size_t)MROWS * NHEADS * 4);
  float*  DA  = (float*)alloc((size_t)MROWS * NHEADS * 4);
  float*  Y   = (float*)alloc((size_t)MROWS * DIN * 4);      // scan output
  bf16_t* Yb  = (bf16_t*)alloc((size_t)MROWS * DIN * 2);     // gated y / gate input reuse
  bf16_t* Wt1 = (bf16_t*)alloc((size_t)PROJP * DIM * 2);     // in_w^T  bf16 (padded)
  bf16_t* Wt2 = (bf16_t*)alloc((size_t)DIM * DIN * 2);       // out_w^T bf16
  bf16_t* Wg  = (bf16_t*)alloc((size_t)DIM * DIN * 2);       // gate_w^T bf16

  const dim3 blk(256);
  k_interleave<<<MROWS * DIM / 256, blk, 0, stream>>>(sem, feat, Hf, Hb);
  k_prep_w384<<<DIM * DIN / 256, blk, 0, stream>>>(gate_w, DIM, Wg);

  // grid sizes for the 4x2-tiled GEMMs (waves = (M/64) * (N/32))
  const int g_in   = (MROWS / (16 * MT)) * (PROJP / (16 * NT)) * 32 / 256;  // 3328
  const int g_out  = (MROWS / (16 * MT)) * (DIM   / (16 * NT)) * 32 / 256;  // 768

  for (int s = 0; s < 2; ++s) {
    float* H = s ? Hb : Hf;
    const Stack& P = st[s];
    for (int L = 0; L < NLAYERS; ++L) {
      k_prep_inw<<<PROJP * DIM / 256, blk, 0, stream>>>(P.in_w + (size_t)L * DIM * PROJ, Wt1);
      k_prep_w384<<<DIM * DIN / 256, blk, 0, stream>>>(P.out_w + (size_t)L * DIN * DIM, DIM, Wt2);
      k_rmsnorm<<<MROWS / 8, blk, 0, stream>>>(H, P.norm_w + L * DIM, Xn);
      k_gemm_inproj<<<g_in, blk, 0, stream>>>(Xn, Wt1, P.in_b + L * PROJ, ZX);
      k_conv<<<MROWS * CONVCH / 256, blk, 0, stream>>>(
          ZX, P.conv_w + L * CONVK * CONVCH, P.conv_b + L * CONVCH, XC);
      k_dt<<<MROWS * NHEADS / 256, blk, 0, stream>>>(
          ZX, P.dt_bias + L * NHEADS, P.A_log + L * NHEADS, DT, DA);
      k_scan<<<BATCH * NHEADS, dim3(512), 0, stream>>>(XC, DT, DA, P.D + L * NHEADS, Y);
      k_ygate<<<MROWS * DIN / 256, blk, 0, stream>>>(Y, ZX, Yb);
      k_gemm_outproj<<<g_out, blk, 0, stream>>>(Yb, Wt2, P.out_b + L * DIM, H);
    }
  }

  k_gateprep<<<MROWS * 2 * DIM / 256, blk, 0, stream>>>(Hf, Hb, Yb);
  k_gemm_gate<<<g_out, blk, 0, stream>>>(Yb, Wg, gate_b, Hf, Hb, (float*)d_out);
}
